// AnchorAttention_48026324303970
// MI455X (gfx1250) — matrix operand order repaired and checked by reading, not compiled, not run on hardware
//
#include <hip/hip_runtime.h>

// ---------------- CDNA5 WMMA types ----------------
typedef __attribute__((ext_vector_type(16))) __bf16 v16bf;
typedef __attribute__((ext_vector_type(8)))  float  v8f;

#define NBATCH 8
#define DIMC   512
#define NTOK   1024
#define NHEADS 8
#define HDIM   64
#define NBH    64           // NBATCH*NHEADS
#define NROWS  8192         // NBATCH*NTOK
#define ATT_SCALE 0.125f    // 64^-0.5

__device__ __forceinline__ unsigned short f32_to_bf16(float f) {
  unsigned u = __float_as_uint(f);
  unsigned r = (u + 0x7FFFu + ((u >> 16) & 1u)) >> 16;
  return (unsigned short)r;
}

__device__ __forceinline__ uint4 pack8_bf16(v8f v) {
  uint4 o;
  o.x = (unsigned)f32_to_bf16(v[0]) | ((unsigned)f32_to_bf16(v[1]) << 16);
  o.y = (unsigned)f32_to_bf16(v[2]) | ((unsigned)f32_to_bf16(v[3]) << 16);
  o.z = (unsigned)f32_to_bf16(v[4]) | ((unsigned)f32_to_bf16(v[5]) << 16);
  o.w = (unsigned)f32_to_bf16(v[6]) | ((unsigned)f32_to_bf16(v[7]) << 16);
  return o;
}

// A-fragment (16x32 bf16, M x K) from row-major [M,K] src (ld in elements).
// ISA layout: lanes 0-15 row M=lane, v0..3 = K 0..7, v4..7 = K 16..23;
// lanes 16-31 same rows, K 8..15 / 24..31 -> two contiguous b128 per lane.
__device__ __forceinline__ v16bf load_frag_a(const __bf16* src, int ld, int row0, int col0) {
  int lane = threadIdx.x & 31;
  int half = lane >> 4;
  int r = row0 + (lane & 15);
  int c = col0 + half * 8;
  union { v16bf v; uint4 u[2]; } u;
  u.u[0] = *(const uint4*)(src + (size_t)r * ld + c);        // K c..c+7
  u.u[1] = *(const uint4*)(src + (size_t)r * ld + c + 16);   // K c+16..c+23
  return u.v;
}

// B-fragment (32x16 bf16, K x N) from B-transposed row-major [N,K] (ld in elems).
// lanes 0-15: col N=lane, K k0..k0+7 / k0+16..k0+23; lanes 16-31: K k0+8..k0+15 / k0+24..k0+31.
__device__ __forceinline__ v16bf load_frag_b(const __bf16* bt, int ld, int n0, int k0) {
  int lane = threadIdx.x & 31;
  int half = lane >> 4;
  int n = n0 + (lane & 15);
  int k = k0 + half * 8;
  const __bf16* p = bt + (size_t)n * ld + k;
  union { v16bf v; uint4 u[2]; } u;
  u.u[0] = *(const uint4*)(p);
  u.u[1] = *(const uint4*)(p + 16);
  return u.v;
}

__device__ __forceinline__ v8f wmma_bf16(v16bf a, v16bf b, v8f c) {
  return __builtin_amdgcn_wmma_f32_16x16x32_bf16(false, a, false, b, (short)0, c, false, false);
}

// ---------------- weight f32 -> bf16 ----------------
__global__ __launch_bounds__(256) void cvt_bf16_kernel(const float* __restrict__ src,
                                                       unsigned short* __restrict__ dst, int n) {
  int i = blockIdx.x * 256 + threadIdx.x;
  if (i < n) dst[i] = f32_to_bf16(src[i]);
}

// ---------------- LayerNorm: x [B][512][1024] (NCHW) -> y [B*1024][512] bf16 token-major
__global__ __launch_bounds__(256) void ln_kernel(const float* __restrict__ x,
                                                 const float* __restrict__ w,
                                                 const float* __restrict__ bb,
                                                 unsigned short* __restrict__ y) {
  __shared__ float tile[16][DIMC + 8];
  int tb = blockIdx.x;                 // 512 tiles: 64 per batch, 16 tokens each
  int b  = tb >> 6;
  int n0 = (tb & 63) * 16;
  const float* xb = x + (size_t)b * DIMC * NTOK;
  for (int idx = threadIdx.x; idx < DIMC * 16; idx += 256) {
    int c = idx >> 4, nn = idx & 15;
    tile[nn][c] = xb[(size_t)c * NTOK + n0 + nn];
  }
  __syncthreads();
  int wid = threadIdx.x >> 5, lane = threadIdx.x & 31;
  #pragma unroll
  for (int t = 0; t < 2; ++t) {
    int nn = wid * 2 + t;
    float v[16];
    float s = 0.f;
    #pragma unroll
    for (int i = 0; i < 16; ++i) { v[i] = tile[nn][lane + 32 * i]; s += v[i]; }
    #pragma unroll
    for (int m = 16; m >= 1; m >>= 1) s += __shfl_xor(s, m, 32);
    float mu = s * (1.f / DIMC);
    float vs = 0.f;
    #pragma unroll
    for (int i = 0; i < 16; ++i) { float d = v[i] - mu; vs += d * d; }
    #pragma unroll
    for (int m = 16; m >= 1; m >>= 1) vs += __shfl_xor(vs, m, 32);
    float rstd = rsqrtf(vs * (1.f / DIMC) + 1e-5f);
    unsigned short* yr = y + ((size_t)(b * NTOK + n0 + nn)) * DIMC;
    #pragma unroll
    for (int i = 0; i < 16; ++i) {
      int c = lane + 32 * i;
      yr[c] = f32_to_bf16((v[i] - mu) * rstd * w[c] + bb[c]);
    }
  }
}

// ---------------- projection GEMM: out[m,o] = sum_k A[m,k]*W[o,k]
// Wave computes 32x64 tile: 2 A-frags x 4 B-frags -> 8 wmma per k-step (2x B reuse).
// A bf16 [8192][512]; W bf16 [512][512] (torch [out,in] == Bt layout).
// mode 0: store [B,H,N,D] bf16 ; mode 1: store [B,H,D,N] bf16 (transposed)
__global__ __launch_bounds__(128) void proj_kernel(const unsigned short* __restrict__ A_,
                                                   const unsigned short* __restrict__ W_,
                                                   unsigned short* __restrict__ out, int mode) {
  int wv = blockIdx.x * 4 + (threadIdx.x >> 5);   // 2048 waves
  int m0 = (wv >> 3) * 32;                        // 256 m-tiles of 32 rows
  int n0 = (wv & 7) * 64;                         // 8 n-tiles of 64
  const __bf16* A = (const __bf16*)A_;
  const __bf16* W = (const __bf16*)W_;
  v8f acc[2][4];
  #pragma unroll
  for (int r = 0; r < 2; ++r)
    #pragma unroll
    for (int j = 0; j < 4; ++j) acc[r][j] = v8f{0,0,0,0,0,0,0,0};
  for (int k0 = 0; k0 < DIMC; k0 += 32) {
    v16bf a0 = load_frag_a(A, DIMC, m0,      k0);
    v16bf a1 = load_frag_a(A, DIMC, m0 + 16, k0);
    v16bf b0 = load_frag_b(W, DIMC, n0,      k0);
    v16bf b1 = load_frag_b(W, DIMC, n0 + 16, k0);
    v16bf b2 = load_frag_b(W, DIMC, n0 + 32, k0);
    v16bf b3 = load_frag_b(W, DIMC, n0 + 48, k0);
    acc[0][0] = wmma_bf16(a0, b0, acc[0][0]);
    acc[0][1] = wmma_bf16(a0, b1, acc[0][1]);
    acc[0][2] = wmma_bf16(a0, b2, acc[0][2]);
    acc[0][3] = wmma_bf16(a0, b3, acc[0][3]);
    acc[1][0] = wmma_bf16(a1, b0, acc[1][0]);
    acc[1][1] = wmma_bf16(a1, b1, acc[1][1]);
    acc[1][2] = wmma_bf16(a1, b2, acc[1][2]);
    acc[1][3] = wmma_bf16(a1, b3, acc[1][3]);
  }
  int lane = threadIdx.x & 31, half = lane >> 4, ln = lane & 15;
  #pragma unroll
  for (int r = 0; r < 2; ++r) {
    int m0r = m0 + r * 16;
    int bb = m0r >> 10, nloc = m0r & 1023;
    #pragma unroll
    for (int j = 0; j < 4; ++j) {
      int o = n0 + j * 16 + ln;
      int h = o >> 6, d = o & 63;
      if (mode == 0) {
        size_t base = ((size_t)(bb * NHEADS + h) * NTOK + nloc) * HDIM + d;
        #pragma unroll
        for (int g = 0; g < 8; ++g)
          out[base + (size_t)(g + 8 * half) * HDIM] = f32_to_bf16(acc[r][j][g]);
      } else {
        size_t base = ((size_t)(bb * NHEADS + h) * HDIM + d) * NTOK + nloc + 8 * half;
        *(uint4*)(out + base) = pack8_bf16(acc[r][j]);
      }
    }
  }
}

// ---------------- flash attention stage (shared by stage1/stage2)
// A  : [BH][1024][64] bf16  (query-like rows)
// Kt : [BH][1024][64] bf16  (Bt layout of key-like: scores = A . Kt^T_as_B)
// Vt : [BH][64][1024] bf16  (value transposed -> Bt layout for P.V)
// mode 0: out = v1^T [BH][64][1024] bf16 ; mode 1: out = [B][1024][512] bf16 (col = h*64+d)
__global__ __launch_bounds__(128) void attn_kernel(const unsigned short* __restrict__ A_,
                                                   const unsigned short* __restrict__ Kt_,
                                                   const unsigned short* __restrict__ Vt_,
                                                   unsigned short* __restrict__ out, int mode) {
  __shared__ unsigned short ldsP[4][16 * 40];     // per-wave 16x32 P tile (padded)
  int widx = threadIdx.x >> 5;
  int wv = blockIdx.x * 4 + widx;                 // 4096 waves
  int bh = wv >> 6;
  int m0 = (wv & 63) * 16;
  const __bf16* A  = (const __bf16*)A_  + (size_t)bh * NTOK * HDIM;
  const __bf16* Kt = (const __bf16*)Kt_ + (size_t)bh * NTOK * HDIM;
  const __bf16* Vt = (const __bf16*)Vt_ + (size_t)bh * HDIM * NTOK;
  unsigned short* ldsw = ldsP[widx];

  v16bf a0 = load_frag_a(A, HDIM, m0, 0);
  v16bf a1 = load_frag_a(A, HDIM, m0, 32);

  float mrun[8], lsum[8];
  v8f o0 = {0,0,0,0,0,0,0,0}, o1 = {0,0,0,0,0,0,0,0},
      o2 = {0,0,0,0,0,0,0,0}, o3 = {0,0,0,0,0,0,0,0};
  #pragma unroll
  for (int g = 0; g < 8; ++g) { mrun[g] = -1e30f; lsum[g] = 0.f; }

  int lane = threadIdx.x & 31, half = lane >> 4, ln = lane & 15;

  for (int j0 = 0; j0 < NTOK; j0 += 32) {
    v16bf kb0 = load_frag_b(Kt, HDIM, j0,      0);
    v16bf kb1 = load_frag_b(Kt, HDIM, j0,      32);
    v16bf kb2 = load_frag_b(Kt, HDIM, j0 + 16, 0);
    v16bf kb3 = load_frag_b(Kt, HDIM, j0 + 16, 32);
    v8f s0 = {0,0,0,0,0,0,0,0}, s1 = {0,0,0,0,0,0,0,0};
    s0 = wmma_bf16(a0, kb0, s0);
    s0 = wmma_bf16(a1, kb1, s0);
    s1 = wmma_bf16(a0, kb2, s1);
    s1 = wmma_bf16(a1, kb3, s1);

    #pragma unroll
    for (int g = 0; g < 8; ++g) {
      float x0 = s0[g] * ATT_SCALE, x1 = s1[g] * ATT_SCALE;
      float rm = fmaxf(x0, x1);
      rm = fmaxf(rm, __shfl_xor(rm, 1, 32));
      rm = fmaxf(rm, __shfl_xor(rm, 2, 32));
      rm = fmaxf(rm, __shfl_xor(rm, 4, 32));
      rm = fmaxf(rm, __shfl_xor(rm, 8, 32));
      float mn = fmaxf(mrun[g], rm);
      float al = __expf(mrun[g] - mn);
      float p0 = __expf(x0 - mn), p1 = __expf(x1 - mn);
      float rs = p0 + p1;
      rs += __shfl_xor(rs, 1, 32);
      rs += __shfl_xor(rs, 2, 32);
      rs += __shfl_xor(rs, 4, 32);
      rs += __shfl_xor(rs, 8, 32);
      lsum[g] = lsum[g] * al + rs;
      mrun[g] = mn;
      o0[g] *= al; o1[g] *= al; o2[g] *= al; o3[g] *= al;
      int row = g + 8 * half;
      ldsw[row * 40 + ln]      = f32_to_bf16(p0);
      ldsw[row * 40 + 16 + ln] = f32_to_bf16(p1);
    }
    asm volatile("s_wait_dscnt 0" ::: "memory");
    v16bf pf  = load_frag_a((const __bf16*)ldsw, 40, 0, 0);
    v16bf vb0 = load_frag_b(Vt, NTOK, 0,  j0);
    v16bf vb1 = load_frag_b(Vt, NTOK, 16, j0);
    v16bf vb2 = load_frag_b(Vt, NTOK, 32, j0);
    v16bf vb3 = load_frag_b(Vt, NTOK, 48, j0);
    o0 = wmma_bf16(pf, vb0, o0);
    o1 = wmma_bf16(pf, vb1, o1);
    o2 = wmma_bf16(pf, vb2, o2);
    o3 = wmma_bf16(pf, vb3, o3);
  }

  #pragma unroll
  for (int g = 0; g < 8; ++g) {
    float inv = 1.f / lsum[g];
    o0[g] *= inv; o1[g] *= inv; o2[g] *= inv; o3[g] *= inv;
  }

  if (mode == 0) {
    v8f* oo[4] = { &o0, &o1, &o2, &o3 };
    #pragma unroll
    for (int t = 0; t < 4; ++t) {
      int d = t * 16 + ln;
      size_t base = ((size_t)bh * HDIM + d) * NTOK + m0 + 8 * half;
      *(uint4*)(out + base) = pack8_bf16(*oo[t]);
    }
  } else {
    int bb = bh >> 3, h = bh & 7;
    v8f* oo[4] = { &o0, &o1, &o2, &o3 };
    #pragma unroll
    for (int t = 0; t < 4; ++t) {
      int col = h * HDIM + t * 16 + ln;
      #pragma unroll
      for (int g = 0; g < 8; ++g) {
        int n = m0 + g + 8 * half;
        out[((size_t)bb * NTOK + n) * DIMC + col] = f32_to_bf16((*oo[t])[g]);
      }
    }
  }
}

// ---------------- final GEMM + bias + residual -> d_out f32 [B][512][1024]
// Same 32x64 wave tile as proj_kernel.
__global__ __launch_bounds__(128) void final_kernel(const unsigned short* __restrict__ A_,
                                                    const unsigned short* __restrict__ W_,
                                                    const float* __restrict__ bias,
                                                    const float* __restrict__ zres,
                                                    float* __restrict__ outp) {
  int wv = blockIdx.x * 4 + (threadIdx.x >> 5);   // 2048 waves
  int m0 = (wv >> 3) * 32;
  int n0 = (wv & 7) * 64;
  const __bf16* A = (const __bf16*)A_;
  const __bf16* W = (const __bf16*)W_;
  v8f acc[2][4];
  #pragma unroll
  for (int r = 0; r < 2; ++r)
    #pragma unroll
    for (int j = 0; j < 4; ++j) acc[r][j] = v8f{0,0,0,0,0,0,0,0};
  for (int k0 = 0; k0 < DIMC; k0 += 32) {
    v16bf a0 = load_frag_a(A, DIMC, m0,      k0);
    v16bf a1 = load_frag_a(A, DIMC, m0 + 16, k0);
    v16bf b0 = load_frag_b(W, DIMC, n0,      k0);
    v16bf b1 = load_frag_b(W, DIMC, n0 + 16, k0);
    v16bf b2 = load_frag_b(W, DIMC, n0 + 32, k0);
    v16bf b3 = load_frag_b(W, DIMC, n0 + 48, k0);
    acc[0][0] = wmma_bf16(a0, b0, acc[0][0]);
    acc[0][1] = wmma_bf16(a0, b1, acc[0][1]);
    acc[0][2] = wmma_bf16(a0, b2, acc[0][2]);
    acc[0][3] = wmma_bf16(a0, b3, acc[0][3]);
    acc[1][0] = wmma_bf16(a1, b0, acc[1][0]);
    acc[1][1] = wmma_bf16(a1, b1, acc[1][1]);
    acc[1][2] = wmma_bf16(a1, b2, acc[1][2]);
    acc[1][3] = wmma_bf16(a1, b3, acc[1][3]);
  }
  int lane = threadIdx.x & 31, half = lane >> 4, ln = lane & 15;
  #pragma unroll
  for (int r = 0; r < 2; ++r) {
    int m0r = m0 + r * 16;
    int bb = m0r >> 10, nloc = m0r & 1023;
    #pragma unroll
    for (int j = 0; j < 4; ++j) {
      int o = n0 + j * 16 + ln;
      float bo = bias[o];
      size_t base = ((size_t)bb * DIMC + o) * NTOK + nloc + 8 * half;
      float4 z0 = *(const float4*)(zres + base);
      float4 z1 = *(const float4*)(zres + base + 4);
      float4 r0, r1;
      r0.x = acc[r][j][0] + bo + z0.x; r0.y = acc[r][j][1] + bo + z0.y;
      r0.z = acc[r][j][2] + bo + z0.z; r0.w = acc[r][j][3] + bo + z0.w;
      r1.x = acc[r][j][4] + bo + z1.x; r1.y = acc[r][j][5] + bo + z1.y;
      r1.z = acc[r][j][6] + bo + z1.z; r1.w = acc[r][j][7] + bo + z1.w;
      *(float4*)(outp + base)     = r0;
      *(float4*)(outp + base + 4) = r1;
    }
  }
}

// ---------------- launch ----------------
extern "C" void kernel_launch(void* const* d_in, const int* in_sizes, int n_in,
                              void* d_out, int out_size, void* d_ws, size_t ws_size,
                              hipStream_t stream) {
  const float* r      = (const float*)d_in[0];
  const float* z      = (const float*)d_in[1];
  const float* nrw    = (const float*)d_in[2];
  const float* nrb    = (const float*)d_in[3];
  const float* nzw    = (const float*)d_in[4];
  const float* nzb    = (const float*)d_in[5];
  const float* bproj  = (const float*)d_in[11];
  float* outp = (float*)d_out;

  char* ws = (char*)d_ws;
  size_t off = 0;
  auto alloc = [&](size_t bytes) -> void* {
    void* p = ws + off;
    off += (bytes + 255) & ~(size_t)255;
    return p;
  };
  unsigned short* Wb[5];
  for (int i = 0; i < 5; ++i) Wb[i] = (unsigned short*)alloc((size_t)DIMC * DIMC * 2);
  unsigned short* rf   = (unsigned short*)alloc((size_t)NROWS * DIMC * 2);
  unsigned short* zf   = (unsigned short*)alloc((size_t)NROWS * DIMC * 2);
  unsigned short* qn   = (unsigned short*)alloc((size_t)NROWS * DIMC * 2);
  unsigned short* kn   = (unsigned short*)alloc((size_t)NROWS * DIMC * 2);
  unsigned short* vT   = (unsigned short*)alloc((size_t)NROWS * DIMC * 2);
  unsigned short* an   = (unsigned short*)alloc((size_t)NROWS * DIMC * 2);
  unsigned short* v1T  = (unsigned short*)alloc((size_t)NROWS * DIMC * 2);
  unsigned short* out2 = (unsigned short*)alloc((size_t)NROWS * DIMC * 2);

  // 1) weights -> bf16 (Wq,Wk,Wv,Wa,Wproj at d_in[6..10]); [out,in] == Bt layout
  for (int i = 0; i < 5; ++i)
    cvt_bf16_kernel<<<dim3(1024), dim3(256), 0, stream>>>((const float*)d_in[6 + i], Wb[i], DIMC * DIMC);

  // 2) LayerNorm (NCHW gather -> token-major bf16)
  ln_kernel<<<dim3(512), dim3(256), 0, stream>>>(r, nrw, nrb, rf);
  ln_kernel<<<dim3(512), dim3(256), 0, stream>>>(z, nzw, nzb, zf);

  // 3) projections: q,k from rf; v,a from zf. v stored transposed [B,H,D,N].
  proj_kernel<<<dim3(512), dim3(128), 0, stream>>>(rf, Wb[0], qn, 0);  // q  [B,H,N,D]
  proj_kernel<<<dim3(512), dim3(128), 0, stream>>>(rf, Wb[1], kn, 0);  // k  [B,H,N,D]
  proj_kernel<<<dim3(512), dim3(128), 0, stream>>>(zf, Wb[2], vT, 1);  // v^T[B,H,D,N]
  proj_kernel<<<dim3(512), dim3(128), 0, stream>>>(zf, Wb[3], an, 0);  // a  [B,H,N,D]

  // 4) stage1: softmax(k a^T * s) v  -> v1^T [B,H,D,N]
  attn_kernel<<<dim3(1024), dim3(128), 0, stream>>>(kn, an, vT, v1T, 0);
  // 5) stage2: softmax(a q^T * s) v1 -> [B,N,512] bf16
  attn_kernel<<<dim3(1024), dim3(128), 0, stream>>>(an, qn, v1T, out2, 1);

  // 6) out2 @ Wproj^T + bproj + z -> [B,512,1024] f32
  final_kernel<<<dim3(512), dim3(128), 0, stream>>>(out2, Wb[4], bproj, z, outp);
}
